// CrossAttention_52948356825344
// MI455X (gfx1250) — matrix operand order repaired and checked by reading, not compiled, not run on hardware
//
#include <hip/hip_runtime.h>
#include <hip/hip_bf16.h>
#include <math.h>

// MI455X (gfx1250) gated cross-attention.
// Memory-bound on the 256MB f32 bias stream (~11us @ 23.3 TB/s); matmuls run
// in bf16 WMMA (v_wmma_f32_16x16x32_bf16) so compute (~13.5 GFLOP) stays far
// under the memory time. q/k/v/gate intermediates (~20MB) live in d_ws and
// stay L2-resident (192MB L2); bias is read with non-temporal policy so the
// one-shot stream does not evict the resident working set.

typedef __attribute__((ext_vector_type(16))) __bf16 v16bf;
typedef __attribute__((ext_vector_type(8)))  __bf16 v8bf;
typedef __attribute__((ext_vector_type(8)))  float  v8f;

#define B_  8
#define S_  1024
#define K_  1024
#define AD  256
#define H_  8
#define CD  32

union V16U { v16bf v; v8bf h[2]; };

static __device__ inline v16bf ld2x8(const __bf16* p0, const __bf16* p1) {
  V16U u;
  u.h[0] = *(const v8bf*)p0;
  u.h[1] = *(const v8bf*)p1;
  return u.v;
}

// ---------------------------------------------------------------------------
// Kernel 1: per-head projections.
//  grid.x = B * (1024/64) row tiles, grid.y = type (0=q,1=k,2=v,3=gate)
//  block  = 256 threads (8 wave32), wave w owns a 16x16 output tile.
// ---------------------------------------------------------------------------
__global__ __launch_bounds__(256)
void proj_kernel(const float* __restrict__ qdata, const float* __restrict__ mdata,
                 const float* __restrict__ qw, const float* __restrict__ qb,
                 const float* __restrict__ kw, const float* __restrict__ vw,
                 const float* __restrict__ gw,
                 __bf16* __restrict__ qbf, __bf16* __restrict__ kbf,
                 __bf16* __restrict__ vbfT, float* __restrict__ gate)
{
  __shared__ __bf16 Xs[64][AD];   // 32 KB activation tile (bf16)
  __shared__ __bf16 WsT[CD][AD];  // 16 KB weight slice, transposed [c][a]
  __shared__ float  Vst[64][CD];  //  8 KB staging for transposed V store

  const int type = blockIdx.y;
  const int b    = blockIdx.x >> 4;
  const int r0   = (blockIdx.x & 15) * 64;
  const int t    = threadIdx.x;
  const int wave = t >> 5, lane = t & 31, lo = lane & 15, hi = lane >> 4;

  const float* X = (type == 1 || type == 2) ? mdata : qdata;
  const float* W = (type == 0) ? qw : (type == 1) ? kw : (type == 2) ? vw : gw;

  // activation tile -> bf16 LDS (row-major, coalesced)
  for (int idx = t; idx < 64 * AD; idx += 256) {
    int r = idx >> 8, c = idx & 255;
    Xs[r][c] = (__bf16)X[(size_t)(b * 1024 + r0 + r) * AD + c];
  }

  const int m0 = (wave >> 1) * 16;   // 4 M-subtiles
  const int n0 = (wave & 1) * 16;    // 2 N-subtiles

  for (int h = 0; h < H_; ++h) {
    __syncthreads();                 // previous head's LDS consumers done
    // weight slice [a][h][c] -> WsT[c][a] bf16
    for (int idx = t; idx < AD * CD; idx += 256) {
      int a = idx >> 5, c = idx & 31;
      WsT[c][a] = (__bf16)W[(a * H_ + h) * CD + c];
    }
    __syncthreads();

    v8f acc = {};
    for (int kk = 0; kk < AD; kk += 32) {
      // A frag 16x32 bf16: lane row = m0+lo, K split by lane-half
      v16bf af = ld2x8(&Xs[m0 + lo][kk + hi * 8],
                       &Xs[m0 + lo][kk + 16 + hi * 8]);
      // B frag 32x16 bf16: lane col = n0+lo, K split by lane-half (same as A)
      v16bf bf = ld2x8(&WsT[n0 + lo][kk + hi * 8],
                       &WsT[n0 + lo][kk + 16 + hi * 8]);
      acc = __builtin_amdgcn_wmma_f32_16x16x32_bf16(false, af, false, bf,
                                                    (short)0, acc, false, false);
    }

    if (type == 2) {
      // stage, then store V transposed: vbfT[b][h][c][key]
      for (int i = 0; i < 8; ++i) Vst[m0 + hi * 8 + i][n0 + lo] = acc[i];
      __syncthreads();
      int c  = t >> 3;
      int kb = (t & 7) * 8;
      v8bf pk;
      for (int j = 0; j < 8; ++j) pk[j] = (__bf16)Vst[kb + j][c];
      __bf16* dst = vbfT + ((size_t)((b * H_ + h) * CD + c) * K_) + r0 + kb;
      *(v8bf*)dst = pk;
    } else {
      for (int i = 0; i < 8; ++i) {
        int row = r0 + m0 + hi * 8 + i;
        int col = n0 + lo;
        size_t o = ((size_t)(b * 1024 + row) * H_ + h) * CD + col;
        float v = acc[i];
        if (type == 0) {
          v = (v + qb[h * CD + col]) * 0.17677669529663687f;  // 1/sqrt(32)
          qbf[o] = (__bf16)v;
        } else if (type == 1) {
          kbf[o] = (__bf16)v;
        } else {
          gate[o] = 1.0f / (1.0f + __expf(-v));
        }
      }
    }
  }
}

// ---------------------------------------------------------------------------
// Kernel 2: flash attention, bias-initialized logits.
//  grid.x = B * (S/128), grid.y = head. 8 waves x 16 query rows each.
//  32-key chunks: logits C-frag starts as bias (single NT-streamed read of the
//  256MB tensor), two WMMAs for QK^T, online softmax with lane-group shuffle
//  reductions, P via LDS (s_wait_dscnt), two WMMAs for PV.
// ---------------------------------------------------------------------------
__global__ __launch_bounds__(256)
void attn_kernel(const float* __restrict__ bias,
                 const __bf16* __restrict__ qbf, const __bf16* __restrict__ kbf,
                 const __bf16* __restrict__ vbfT, const float* __restrict__ gate,
                 float* __restrict__ out)
{
  __shared__ __bf16 pbuf[8][16][CD];  // 8 KB: per-wave P tile

  const int b    = blockIdx.x >> 3;
  const int q0   = (blockIdx.x & 7) * 128;
  const int h    = blockIdx.y;
  const int t    = threadIdx.x;
  const int wave = t >> 5, lane = t & 31, lo = lane & 15, hi = lane >> 4;
  const int qrow = q0 + wave * 16;

  // Q A-fragment: constant across the K loop (head dim 32 == one bf16 K-step)
  const __bf16* qrowp = qbf + ((size_t)(b * S_ + qrow + lo) * H_ + h) * CD;
  v16bf aq = ld2x8(qrowp + hi * 8, qrowp + 16 + hi * 8);

  const float*  biasbh  = bias + (size_t)(b * H_ + h) * S_ * K_;
  const __bf16* kb_base = kbf  + ((size_t)b * K_ * H_ + h) * CD;       // + key*H_*CD
  const __bf16* vt_base = vbfT + (size_t)((b * H_ + h) * CD) * K_;     // + c*K_ + key

  v8f acc0 = {}, acc1 = {};
  float mrow[8], lrow[8];
  for (int i = 0; i < 8; ++i) { mrow[i] = -3.0e38f; lrow[i] = 0.0f; }

  for (int kc = 0; kc < K_; kc += 32) {
    // C frags initialized with bias -> logits = q.k + bias in one WMMA.
    // Non-temporal: 256MB one-shot stream must not evict L2-resident k/v/q/gate.
    v8f c0, c1;
    for (int i = 0; i < 8; ++i) {
      size_t ro = (size_t)(qrow + hi * 8 + i) * K_ + kc;
      c0[i] = __builtin_nontemporal_load(&biasbh[ro + lo]);
      c1[i] = __builtin_nontemporal_load(&biasbh[ro + 16 + lo]);
      if (kc + 32 < K_) __builtin_prefetch(&biasbh[ro + 32], 0, 1);  // global_prefetch_b8
    }
    const __bf16* k0p = kb_base + (size_t)(kc + lo) * H_ * CD + hi * 8;
    const __bf16* k1p = kb_base + (size_t)(kc + 16 + lo) * H_ * CD + hi * 8;
    v16bf kb0 = ld2x8(k0p, k0p + 16);
    v16bf kb1 = ld2x8(k1p, k1p + 16);
    c0 = __builtin_amdgcn_wmma_f32_16x16x32_bf16(false, aq, false, kb0, (short)0, c0, false, false);
    c1 = __builtin_amdgcn_wmma_f32_16x16x32_bf16(false, aq, false, kb1, (short)0, c1, false, false);

    // online softmax over this 32-key chunk
    float p0[8], p1[8];
    for (int i = 0; i < 8; ++i) {
      float mx = fmaxf(c0[i], c1[i]);
      for (int m = 1; m < 16; m <<= 1) mx = fmaxf(mx, __shfl_xor(mx, m, 32));
      float mn = fmaxf(mrow[i], mx);
      float sc = __expf(mrow[i] - mn);
      p0[i] = __expf(c0[i] - mn);
      p1[i] = __expf(c1[i] - mn);
      float rs = p0[i] + p1[i];
      for (int m = 1; m < 16; m <<= 1) rs += __shfl_xor(rs, m, 32);
      lrow[i] = lrow[i] * sc + rs;
      mrow[i] = mn;
      acc0[i] *= sc;
      acc1[i] *= sc;
    }

    // C-layout -> A-layout via LDS (same-wave RAW: split DS counter wait only)
    for (int i = 0; i < 8; ++i) {
      pbuf[wave][hi * 8 + i][lo]      = (__bf16)p0[i];
      pbuf[wave][hi * 8 + i][16 + lo] = (__bf16)p1[i];
    }
    asm volatile("s_wait_dscnt 0" ::: "memory");

    v16bf pa = ld2x8(&pbuf[wave][lo][hi * 8], &pbuf[wave][lo][16 + hi * 8]);
    const __bf16* v0p = vt_base + (size_t)lo * K_ + kc + hi * 8;
    const __bf16* v1p = vt_base + (size_t)(16 + lo) * K_ + kc + hi * 8;
    v16bf vb0 = ld2x8(v0p, v0p + 16);
    v16bf vb1 = ld2x8(v1p, v1p + 16);
    acc0 = __builtin_amdgcn_wmma_f32_16x16x32_bf16(false, pa, false, vb0, (short)0, acc0, false, false);
    acc1 = __builtin_amdgcn_wmma_f32_16x16x32_bf16(false, pa, false, vb1, (short)0, acc1, false, false);
  }

  // normalize + gate, f32 output [B,S,H,32]
  for (int i = 0; i < 8; ++i) {
    int row = qrow + hi * 8 + i;
    float inv = 1.0f / lrow[i];
    size_t o = ((size_t)(b * S_ + row) * H_ + h) * CD;
    out[o + lo]      = acc0[i] * inv * gate[o + lo];
    out[o + 16 + lo] = acc1[i] * inv * gate[o + 16 + lo];
  }
}

// ---------------------------------------------------------------------------
extern "C" void kernel_launch(void* const* d_in, const int* in_sizes, int n_in,
                              void* d_out, int out_size, void* d_ws, size_t ws_size,
                              hipStream_t stream) {
  const float* q_data = (const float*)d_in[0];
  const float* m_data = (const float*)d_in[1];
  const float* bias   = (const float*)d_in[2];
  const float* qw     = (const float*)d_in[3];
  const float* qb     = (const float*)d_in[4];
  const float* kw     = (const float*)d_in[5];
  const float* vw     = (const float*)d_in[6];
  const float* gw     = (const float*)d_in[7];

  char* ws = (char*)d_ws;                       // 20 MB total
  __bf16* qbf  = (__bf16*)(ws);                 // 4 MB  [B,S,H,32] bf16
  __bf16* kbf  = (__bf16*)(ws + (4  << 20));    // 4 MB  [B,K,H,32] bf16
  __bf16* vbfT = (__bf16*)(ws + (8  << 20));    // 4 MB  [B,H,32,K] bf16 (transposed)
  float*  gate = (float*) (ws + (12 << 20));    // 8 MB  [B,S,H,32] f32

  dim3 g1(B_ * (S_ / 64), 4), b1(256);
  proj_kernel<<<g1, b1, 0, stream>>>(q_data, m_data, qw, qb, kw, vw, gw,
                                     qbf, kbf, vbfT, gate);

  dim3 g2(B_ * (S_ / 128), H_), b2(256);
  attn_kernel<<<g2, b2, 0, stream>>>(bias, qbf, kbf, vbfT, gate, (float*)d_out);
}
